// SpikingParrot_40252433498552
// MI455X (gfx1250) — matrix operand order repaired and checked by reading, not compiled, not run on hardware
//
// SpikingParrot for MI455X (gfx1250, wave32, WMMA).
// Strategy:
//  - All GEMMs run on v_wmma_f32_16x16x32_bf16 (bf16 inputs, f32 accumulate).
//  - Weights converted once to bf16 in workspace (halves HBM/L2 bytes; whh
//    [4096x1024] = 8MB bf16 lives in the 192MB L2 across the 256-step scans).
//  - Generic GEMM blocks 4 M-tiles per wave (MT=4) so each weight fragment
//    feeds 4 WMMAs: the big GEMMs (Xg precompute, lin/mix, 134-GFLOP fc
//    logits GEMM) are weight-bandwidth bound and this cuts weight bytes/flop 4x.
//  - Batch (8) padded to 16 rows = one WMMA M-tile; recurrent mem@whh^T per
//    step is a 16x4096 WMMA GEMM fused with the LSTM pointwise via LDS.
//  - Time-parallel work hoisted out of the scans as large WMMA GEMMs
//    (x@wih^T for all t, lin/mix over full sequences, fc logits GEMM with a
//    (b,t) row-remap epilogue writing d_out directly).
//  - concat([o,x])@mix^T done as two accumulate-GEMMs on column slices of
//    mix_w; DyT fused into GEMM epilogues.
#include <hip/hip_runtime.h>
#include <hip/hip_bf16.h>
#include <math.h>

#define B_    8
#define SRC_  256
#define TGT_  128
#define V_    32000
#define E_    512
#define H_    1024
#define G4_   4096   // 4*H

typedef __bf16 bf16;
typedef __attribute__((ext_vector_type(16))) __bf16 v16bf;
typedef __attribute__((ext_vector_type(8)))  float  v8f;

__device__ __forceinline__ float sigmoidf_(float x) { return 1.f / (1.f + __expf(-x)); }

// ---------------------------------------------------------------------------
// WMMA fragment layouts per CDNA5 ISA (wave32).
// A (16x32 bf16, MxK): lane l<16 -> M=l, K chunks [0..7] & [16..23];
//                      lane l>=16 -> M=l-16, K chunks [8..15] & [24..31].
// B (32x16 bf16, KxN): lane l -> N=l&15; same K chunks as A (lanes 0-15 K [0..7]&[16..23], lanes 16-31 K [8..15]&[24..31]).
// Since C = A * W^T with W stored [N,K] row-major, W rows feed B directly.
// C/D: VGPR r holds M = r + 8*(lane>>4), N = lane&15.
// ---------------------------------------------------------------------------

// ---------------- generic GEMM:  C[M,N] (+)= A[M,K] * W[N,K]^T -------------
// MT = number of 16-row M-tiles per wave (weight-fragment reuse factor).
// flags: 1 = accumulate (read C), 2 = DyT epilogue, 4 = (b,t) row-remap to out
template <int MT>
__global__ __launch_bounds__(256) void k_gemm(
    float* __restrict__ C, bf16* __restrict__ Cbf,
    const bf16* __restrict__ A, const bf16* __restrict__ W,
    const float* __restrict__ bias,
    int M, int N, int K,
    int lda, int ldw, int ldc, int ldcb, int wcol0, int flags,
    const float* __restrict__ dya, const float* __restrict__ dyw,
    const float* __restrict__ dyb, int remapS)
{
  const int lane = threadIdx.x & 31;            // wave32
  const int wv   = threadIdx.x >> 5;            // 8 waves / block
  const int col0 = (blockIdx.x * 8 + wv) * 16;
  const int row0 = blockIdx.y * (16 * MT);
  if (col0 >= N) return;                        // wave-uniform

  const int lm = lane & 15;
  const int ka = (lane >> 4) * 8;
  const int kb = (lane >> 4) * 8;
  const bf16* pa = A + (size_t)(row0 + lm) * lda + ka;
  const bf16* pw = W + (size_t)(col0 + lm) * ldw + wcol0 + kb;

  v8f acc[MT];
#pragma unroll
  for (int i = 0; i < MT; ++i) acc[i] = (v8f){};

  for (int k0 = 0; k0 < K; k0 += 32) {
    v16bf b;                                    // weight fragment: loaded once,
    ((uint4*)&b)[0] = *(const uint4*)(pw + k0); // reused by MT WMMAs
    ((uint4*)&b)[1] = *(const uint4*)(pw + k0 + 16);
    if (k0 + 256 < K) {                         // lowers to global_prefetch_b8
      __builtin_prefetch(pa + k0 + 256, 0, 1);
      __builtin_prefetch(pw + k0 + 256, 0, 1);
    }
#pragma unroll
    for (int i = 0; i < MT; ++i) {
      v16bf a;
      const bf16* pai = pa + (size_t)i * 16 * lda + k0;
      ((uint4*)&a)[0] = *(const uint4*)(pai);
      ((uint4*)&a)[1] = *(const uint4*)(pai + 16);
      acc[i] = __builtin_amdgcn_wmma_f32_16x16x32_bf16(false, a, false, b,
                                                       (short)0, acc[i],
                                                       false, false);
    }
  }

  // epilogue
  const int n  = lane & 15;
  const int mb = (lane >> 4) * 8;
  const int cn = col0 + n;
  const float bv = bias ? bias[cn] : 0.f;
  float al = 0.f, dw = 0.f, db = 0.f;
  if (flags & 2) { al = dya[0]; dw = dyw[cn]; db = dyb[cn]; }
#pragma unroll
  for (int i = 0; i < MT; ++i) {
#pragma unroll
    for (int r = 0; r < 8; ++r) {
      const int mm = row0 + i * 16 + mb + r;
      float v = acc[i][r] + bv;
      if (flags & 1) v += C[(size_t)mm * ldc + cn];
      if (flags & 2) v = dw * tanhf(al * v) + db;
      if (flags & 4) {
        const int bb = mm & 15, tt = mm >> 4;   // rows are t*16 + b
        if (bb < B_) C[((size_t)bb * remapS + tt) * (size_t)ldc + cn] = v;
      } else {
        C[(size_t)mm * ldc + cn] = v;
        if (Cbf) Cbf[(size_t)mm * ldcb + cn] = (bf16)v;
      }
    }
  }
}

// ------------- fused SLSTM cell step (one timestep, padded batch 16) -------
// gates[16,4H] = (Xg[t] precomputed x@wih^T + biases  |  x@wih^T + gbias)
//               + mem@whh^T   (WMMA), then i,f,g,o pointwise via LDS.
__global__ __launch_bounds__(256) void k_slstm_step(
    const float* __restrict__ Xg,        // [16,4H] slice incl. biases, or null
    const bf16*  __restrict__ x,         // [16,in_d] dynamic input (if Xg null)
    const bf16*  __restrict__ wih, int in_d,
    const bf16*  __restrict__ whh,
    const float* __restrict__ gbias,     // bih+bhh (used when Xg null)
    const bf16*  __restrict__ mem_in,    // [16,H] bf16
    const float* __restrict__ syn_in,    // [16,H]
    float* __restrict__ syn_out,
    float* __restrict__ mem_out,         // f32 (seq slice or scratch)
    bf16*  __restrict__ mem_bf_out,
    bf16*  __restrict__ spk_out,         // [16,H] bf16 spikes
    const float* __restrict__ thr)
{
  __shared__ float gl[4][16][32];
  const int lane = threadIdx.x & 31;
  const int wv   = threadIdx.x >> 5;
  const int g    = wv >> 1;                       // gate: 0=i 1=f 2=g 3=o
  const int sub  = wv & 1;
  const int col  = g * H_ + blockIdx.x * 32 + sub * 16;
  const int lm   = lane & 15;
  const int ka   = (lane >> 4) * 8;
  const int kb   = (lane >> 4) * 8;

  v8f acc = {};
  if (x) {                                        // dynamic x @ wih^T
    const bf16* pa = x + (size_t)lm * in_d + ka;
    const bf16* pb = wih + (size_t)(col + lm) * in_d + kb;
    for (int k0 = 0; k0 < in_d; k0 += 32) {
      v16bf a, b;
      ((uint4*)&a)[0] = *(const uint4*)(pa + k0);
      ((uint4*)&a)[1] = *(const uint4*)(pa + k0 + 16);
      ((uint4*)&b)[0] = *(const uint4*)(pb + k0);
      ((uint4*)&b)[1] = *(const uint4*)(pb + k0 + 16);
      acc = __builtin_amdgcn_wmma_f32_16x16x32_bf16(false, a, false, b,
                                                    (short)0, acc, false, false);
    }
  }
  {                                               // mem @ whh^T
    const bf16* pa = mem_in + (size_t)lm * H_ + ka;
    const bf16* pb = whh + (size_t)(col + lm) * H_ + kb;
    for (int k0 = 0; k0 < H_; k0 += 32) {
      v16bf a, b;
      ((uint4*)&a)[0] = *(const uint4*)(pa + k0);
      ((uint4*)&a)[1] = *(const uint4*)(pa + k0 + 16);
      ((uint4*)&b)[0] = *(const uint4*)(pb + k0);
      ((uint4*)&b)[1] = *(const uint4*)(pb + k0 + 16);
      acc = __builtin_amdgcn_wmma_f32_16x16x32_bf16(false, a, false, b,
                                                    (short)0, acc, false, false);
    }
  }
  const int n  = lane & 15;
  const int mb = (lane >> 4) * 8;
  const float badd = Xg ? 0.f : gbias[col + n];
#pragma unroll
  for (int r = 0; r < 8; ++r) {
    const int mm = mb + r;
    float v = acc[r] + badd;
    if (Xg) v += Xg[(size_t)mm * G4_ + col + n];
    gl[g][mm][sub * 16 + n] = v;
  }
  __syncthreads();

  const float th = thr[0];
  for (int e = threadIdx.x; e < 512; e += 256) {  // 16 rows x 32 h
    const int mm = e >> 5, hl = e & 31;
    const int h = blockIdx.x * 32 + hl;
    const float gi = gl[0][mm][hl], gf = gl[1][mm][hl];
    const float gG = gl[2][mm][hl], go = gl[3][mm][hl];
    const float ns = sigmoidf_(gf) * syn_in[(size_t)mm * H_ + h]
                   + sigmoidf_(gi) * tanhf(gG);
    const float nm = sigmoidf_(go) * tanhf(ns);
    syn_out[(size_t)mm * H_ + h]    = ns;
    mem_out[(size_t)mm * H_ + h]    = nm;
    mem_bf_out[(size_t)mm * H_ + h] = (bf16)nm;
    spk_out[(size_t)mm * H_ + h]    = (bf16)(((nm - th) > 0.f) ? 1.f : 0.f);
  }
}

// ------------------------- small helper kernels ----------------------------
__global__ __launch_bounds__(256) void k_cast_bf16(bf16* d, const float* s, long n) {
  long i = ((long)blockIdx.x * 256 + threadIdx.x) * 4;
  if (i + 3 < n) {
    const float4 v = *(const float4*)(s + i);
    d[i] = (bf16)v.x; d[i+1] = (bf16)v.y; d[i+2] = (bf16)v.z; d[i+3] = (bf16)v.w;
  } else {
    for (long j = i; j < n; ++j) d[j] = (bf16)s[j];
  }
}

__global__ __launch_bounds__(256) void k_addvec(float* d, const float* a,
                                                const float* b, int n) {
  int i = blockIdx.x * 256 + threadIdx.x;
  if (i < n) d[i] = a[i] + b[i];
}

// embed + DyT, time-major padded [S][16][E]; pad rows (b>=8) zeroed
__global__ __launch_bounds__(256) void k_embed_dyt(
    bf16* __restrict__ out, const float* __restrict__ emb, const int* __restrict__ idx,
    const float* __restrict__ alpha, const float* __restrict__ w,
    const float* __restrict__ b, int S, int E)
{
  const int s = blockIdx.x, m = blockIdx.y;
  bf16* d = out + ((size_t)s * 16 + m) * E;
  if (m >= B_) { for (int e = threadIdx.x; e < E; e += 256) d[e] = (bf16)0.f; return; }
  const int tok = idx[m * S + s];
  const float* ep = emb + (size_t)tok * E;
  const float al = alpha[0];
  for (int e = threadIdx.x; e < E; e += 256)
    d[e] = (bf16)(w[e] * tanhf(al * ep[e]) + b[e]);
}

__global__ __launch_bounds__(256) void k_rev_time(bf16* dst, const bf16* src,
                                                  int S, int D) {
  const int t = blockIdx.x, m = blockIdx.y;
  const bf16* s = src + ((size_t)(S - 1 - t) * 16 + m) * D;
  bf16* d = dst + ((size_t)t * 16 + m) * D;
  for (int e = threadIdx.x; e < D; e += 256) d[e] = s[e];
}

// out2[t] = [ f[t] , b[S-1-t] ]  along feature dim
__global__ __launch_bounds__(256) void k_concat_fb(bf16* out2, const bf16* f,
                                                   const bf16* bk, int S, int D) {
  const int t = blockIdx.x, m = blockIdx.y;
  bf16* d = out2 + ((size_t)t * 16 + m) * (2 * D);
  const bf16* pf = f + ((size_t)t * 16 + m) * D;
  const bf16* pb = bk + ((size_t)(S - 1 - t) * 16 + m) * D;
  for (int e = threadIdx.x; e < D; e += 256) { d[e] = pf[e]; d[D + e] = pb[e]; }
}

// dot-product attention + softmax + ctx + DyT; one block per (padded) batch row
__global__ __launch_bounds__(256) void k_attention(
    bf16* __restrict__ ctx_out,        // row stride 2H, write [0,H) at this ptr
    const float* __restrict__ q,       // [16,H]
    const float* __restrict__ memseq,  // [SRC*16,H] encoder mems
    const float* __restrict__ aa, const float* __restrict__ aw,
    const float* __restrict__ ab)
{
  const int b = blockIdx.x;
  const int tid = threadIdx.x;
  if (b >= B_) {                        // zero pad rows (keeps xcur deterministic)
    for (int h = tid; h < H_; h += 256) ctx_out[(size_t)b * (2 * H_) + h] = (bf16)0.f;
    return;
  }
  __shared__ float sc[SRC_];
  __shared__ float red[256];
  const float* qp = q + (size_t)b * H_;
  {
    const float* mp = memseq + ((size_t)tid * 16 + b) * H_;
    float a0 = 0.f;
    for (int h = 0; h < H_; ++h) a0 += mp[h] * qp[h];
    sc[tid] = a0;
  }
  __syncthreads();
  red[tid] = sc[tid]; __syncthreads();
  for (int o = 128; o > 0; o >>= 1) {
    if (tid < o) red[tid] = fmaxf(red[tid], red[tid + o]);
    __syncthreads();
  }
  const float mx = red[0]; __syncthreads();
  const float ex = __expf(sc[tid] - mx);
  red[tid] = ex; __syncthreads();
  for (int o = 128; o > 0; o >>= 1) {
    if (tid < o) red[tid] += red[tid + o];
    __syncthreads();
  }
  const float inv = 1.f / red[0]; __syncthreads();
  sc[tid] = ex * inv;
  __syncthreads();
  const float al = aa[0];
  for (int h = tid; h < H_; h += 256) {
    float a0 = 0.f;
    for (int s = 0; s < SRC_; ++s) a0 += sc[s] * memseq[((size_t)s * 16 + b) * H_ + h];
    ctx_out[(size_t)b * (2 * H_) + h] = (bf16)(aw[h] * tanhf(al * a0) + ab[h]);
  }
}

// ------------------------------ host side ----------------------------------
struct SLayer {
  const float *lin_b,*lin_w,*mix_b,*mix_w,*na,*nb,*nw,*bhh,*bih,*thr,*whh,*wih;
};
struct DLayer { SLayer s; const float *an_a,*an_b,*an_w,*attn; };

static SLayer slayer_at(void* const* din, int base) {
  SLayer L;
  L.lin_b=(const float*)din[base+0];  L.lin_w=(const float*)din[base+1];
  L.mix_b=(const float*)din[base+2];  L.mix_w=(const float*)din[base+3];
  L.na=(const float*)din[base+4];     L.nb=(const float*)din[base+5];
  L.nw=(const float*)din[base+6];     L.bhh=(const float*)din[base+7];
  L.bih=(const float*)din[base+8];    L.thr=(const float*)din[base+9];
  L.whh=(const float*)din[base+10];   L.wih=(const float*)din[base+11];
  return L;
}
static DLayer dlayer_at(void* const* din, int base) {
  DLayer D;
  D.an_a=(const float*)din[base+0];   D.an_b=(const float*)din[base+1];
  D.an_w=(const float*)din[base+2];   D.attn=(const float*)din[base+3];
  D.s.lin_b=(const float*)din[base+4];D.s.lin_w=(const float*)din[base+5];
  D.s.mix_b=(const float*)din[base+6];D.s.mix_w=(const float*)din[base+7];
  D.s.na=(const float*)din[base+8];   D.s.nb=(const float*)din[base+9];
  D.s.nw=(const float*)din[base+10];  D.s.bhh=(const float*)din[base+11];
  D.s.bih=(const float*)din[base+12]; D.s.thr=(const float*)din[base+13];
  D.s.whh=(const float*)din[base+14]; D.s.wih=(const float*)din[base+15];
  return D;
}

struct Bump {
  char* p; size_t off;
  void* take(size_t bytes) {
    void* r = p + off;
    off = (off + bytes + 255) & ~(size_t)255;
    return r;
  }
};

static inline void launch_gemm(hipStream_t st, float* C, bf16* Cbf,
    const bf16* A, const bf16* W, const float* bias,
    int M, int N, int K, int lda, int ldw, int ldc, int ldcb, int wcol0,
    int flags, const float* dya, const float* dyw, const float* dyb, int remapS)
{
  if ((M & 63) == 0) {   // 4 M-tiles per wave: 4x weight-fragment reuse
    dim3 g((unsigned)((N + 127) / 128), (unsigned)(M / 64));
    k_gemm<4><<<g, 256, 0, st>>>(C, Cbf, A, W, bias, M, N, K, lda, ldw, ldc,
                                 ldcb, wcol0, flags, dya, dyw, dyb, remapS);
  } else {
    dim3 g((unsigned)((N + 127) / 128), (unsigned)(M / 16));
    k_gemm<1><<<g, 256, 0, st>>>(C, Cbf, A, W, bias, M, N, K, lda, ldw, ldc,
                                 ldcb, wcol0, flags, dya, dyw, dyb, remapS);
  }
}

static int run_cell_seq(hipStream_t st, const float* Xg, const bf16* whh,
                        const float* thr, int S,
                        float* syn0, float* syn1, bf16* mem0, bf16* mem1,
                        float* memseq, float* memscr, bf16* spkseq)
{
  float* synP[2] = { syn0, syn1 };
  bf16*  memP[2] = { mem0, mem1 };
  int cur = 0;
  for (int t = 0; t < S; ++t) {
    float* mo = memseq ? (memseq + (size_t)t * 16 * H_) : memscr;
    k_slstm_step<<<dim3(H_ / 32), 256, 0, st>>>(
        Xg + (size_t)t * 16 * G4_, nullptr, nullptr, 0,
        whh, nullptr, memP[cur], synP[cur], synP[cur ^ 1], mo, memP[cur ^ 1],
        spkseq + (size_t)t * 16 * H_, thr);
    cur ^= 1;
  }
  return cur;   // current state lives in synP[cur]/memP[cur]
}

extern "C" void kernel_launch(void* const* d_in, const int* in_sizes, int n_in,
                              void* d_out, int out_size, void* d_ws, size_t ws_size,
                              hipStream_t stream)
{
  (void)in_sizes; (void)n_in; (void)out_size;
  // ---- input leaf map: jax tree flattening (dicts in sorted-key order) ----
  // 0 dec_emb | 1..16 dec_layers[0] (attn_norm.{alpha,b,w}, attn_w, lin_b,
  //   lin_w, mix_b, mix_w, norm.{alpha,b,w}, slstm.{bhh,bih,thr,whh,wih})
  // 17..32 dec_layers[1] | 33..35 dec_norm{alpha,b,w} | 36..47 enc_b
  // 48 enc_emb | 49..60 enc_f | 61..72 enc_layers[0] | 73..84 enc_layers[1]
  // 85..87 enc_norm | 88 fc_b | 89 fc_w | 90 src | 91 tgt
  const float* dec_emb = (const float*)d_in[0];
  DLayer D0 = dlayer_at(d_in, 1);
  DLayer D1 = dlayer_at(d_in, 17);
  const float* dn_a = (const float*)d_in[33];
  const float* dn_b = (const float*)d_in[34];
  const float* dn_w = (const float*)d_in[35];
  SLayer EB = slayer_at(d_in, 36);
  const float* enc_emb = (const float*)d_in[48];
  SLayer EF = slayer_at(d_in, 49);
  SLayer L0 = slayer_at(d_in, 61);
  SLayer L1 = slayer_at(d_in, 73);
  const float* en_a = (const float*)d_in[85];
  const float* en_b = (const float*)d_in[86];
  const float* en_w = (const float*)d_in[87];
  const float* fc_b = (const float*)d_in[88];
  const float* fc_w = (const float*)d_in[89];
  const int* src = (const int*)d_in[90];
  const int* tgt = (const int*)d_in[91];
  float* out = (float*)d_out;

  Bump ws{ (char*)d_ws, 0 };
  auto BF = [&](size_t n) { return (bf16*)ws.take(n * sizeof(bf16)); };
  auto F32 = [&](size_t n) { return (float*)ws.take(n * sizeof(float)); };

  // ---- bf16 weight mirrors ----
  bf16 *wih_ef = BF(4096ull*512),  *whh_ef = BF(4096ull*1024);
  bf16 *lin_ef = BF(1024ull*1024), *mix_ef = BF(1024ull*1536);
  bf16 *wih_eb = BF(4096ull*512),  *whh_eb = BF(4096ull*1024);
  bf16 *lin_eb = BF(1024ull*1024), *mix_eb = BF(1024ull*1536);
  bf16 *wih_l0 = BF(4096ull*2048), *whh_l0 = BF(4096ull*1024);
  bf16 *lin_l0 = BF(1024ull*1024), *mix_l0 = BF(1024ull*3072);
  bf16 *wih_l1 = BF(4096ull*1024), *whh_l1 = BF(4096ull*1024); // lin/mix of L1 unused
  bf16 *wih_d0 = BF(4096ull*512),  *whh_d0 = BF(4096ull*1024);
  bf16 *lin_d0 = BF(1024ull*1024), *mix_d0 = BF(1024ull*1536), *attn_d0 = BF(1024ull*1024);
  bf16 *wih_d1 = BF(4096ull*2048), *whh_d1 = BF(4096ull*1024);
  bf16 *lin_d1 = BF(1024ull*1024), *mix_d1 = BF(1024ull*3072), *attn_d1 = BF(1024ull*1024);
  bf16 *fc_bf  = BF((size_t)V_ * 2048);
  float *gb_ef = F32(G4_), *gb_eb = F32(G4_), *gb_l0 = F32(G4_);
  float *gb_l1 = F32(G4_), *gb_d0 = F32(G4_), *gb_d1 = F32(G4_);

  // ---- activations ----
  float* Xg     = F32(4096ull * G4_);          // reused per sequence (also dec0)
  bf16* x_enc   = BF(4096ull * E_);
  bf16* x_rev   = BF(4096ull * E_);
  bf16* tx      = BF(2048ull * E_);
  bf16* spkseq  = BF(4096ull * H_);
  float* linout = F32(4096ull * H_);
  bf16* linbf   = BF(4096ull * H_);
  float* mixbig = F32(4096ull * H_);
  bf16* fout    = BF(4096ull * H_);
  bf16* bout    = BF(4096ull * H_);
  bf16* cat2h   = BF(4096ull * 2 * H_);
  bf16* l0out   = BF(4096ull * H_);
  float* mseq0  = F32(4096ull * H_);           // enc layer mems (attention)
  float* mseq1  = F32(4096ull * H_);
  float *synT0 = F32(16*H_), *synT1 = F32(16*H_);
  float *synA0 = F32(16*H_), *synA1 = F32(16*H_);
  float *synB0 = F32(16*H_), *synB1 = F32(16*H_);
  bf16 *memT0 = BF(16*H_), *memT1 = BF(16*H_);
  bf16 *memA0 = BF(16*H_), *memA1 = BF(16*H_);
  bf16 *memB0 = BF(16*H_), *memB1 = BF(16*H_);
  float* memscr = F32(16*H_);
  bf16* decout  = BF(2048ull * 2 * H_);
  bf16* spk_st  = BF(16*H_);
  float* o_f    = F32(16*H_);
  bf16* o_bf    = BF(16*H_);
  float* mixtmp = F32(16*H_);
  float* q_f    = F32(16*H_);
  bf16* xcur0   = BF(16 * 2 * H_);
  (void)ws_size;   // assumed large enough (~470 MB high-water)

  auto cast = [&](bf16* dst, const float* s, long n) {
    k_cast_bf16<<<dim3((unsigned)((n + 1023) / 1024)), 256, 0, stream>>>(dst, s, n);
  };
  auto addv = [&](float* d, const float* a, const float* b) {
    k_addvec<<<dim3(G4_ / 256), 256, 0, stream>>>(d, a, b, G4_);
  };

  // ---- weight conversion + gate-bias precompute ----
  cast(wih_ef, EF.wih, 4096l*512);  cast(whh_ef, EF.whh, 4096l*1024);
  cast(lin_ef, EF.lin_w, 1024l*1024); cast(mix_ef, EF.mix_w, 1024l*1536);
  cast(wih_eb, EB.wih, 4096l*512);  cast(whh_eb, EB.whh, 4096l*1024);
  cast(lin_eb, EB.lin_w, 1024l*1024); cast(mix_eb, EB.mix_w, 1024l*1536);
  cast(wih_l0, L0.wih, 4096l*2048); cast(whh_l0, L0.whh, 4096l*1024);
  cast(lin_l0, L0.lin_w, 1024l*1024); cast(mix_l0, L0.mix_w, 1024l*3072);
  cast(wih_l1, L1.wih, 4096l*1024); cast(whh_l1, L1.whh, 4096l*1024);
  cast(wih_d0, D0.s.wih, 4096l*512);  cast(whh_d0, D0.s.whh, 4096l*1024);
  cast(lin_d0, D0.s.lin_w, 1024l*1024); cast(mix_d0, D0.s.mix_w, 1024l*1536);
  cast(attn_d0, D0.attn, 1024l*1024);
  cast(wih_d1, D1.s.wih, 4096l*2048); cast(whh_d1, D1.s.whh, 4096l*1024);
  cast(lin_d1, D1.s.lin_w, 1024l*1024); cast(mix_d1, D1.s.mix_w, 1024l*3072);
  cast(attn_d1, D1.attn, 1024l*1024);
  cast(fc_bf, fc_w, (long)V_ * 2048);
  addv(gb_ef, EF.bih, EF.bhh); addv(gb_eb, EB.bih, EB.bhh);
  addv(gb_l0, L0.bih, L0.bhh); addv(gb_l1, L1.bih, L1.bhh);
  addv(gb_d0, D0.s.bih, D0.s.bhh); addv(gb_d1, D1.s.bih, D1.s.bhh);

  // ================================ encoder ================================
  k_embed_dyt<<<dim3(SRC_, 16), 256, 0, stream>>>(x_enc, enc_emb, src, en_a, en_w, en_b, SRC_, E_);
  k_rev_time<<<dim3(SRC_, 16), 256, 0, stream>>>(x_rev, x_enc, SRC_, E_);

  // --- enc_f ---
  launch_gemm(stream, Xg, nullptr, x_enc, wih_ef, gb_ef,
              4096, G4_, E_, E_, E_, G4_, 0, 0, 0, nullptr, nullptr, nullptr, 0);
  hipMemsetAsync(synT0, 0, 16 * H_ * sizeof(float), stream);
  hipMemsetAsync(memT0, 0, 16 * H_ * sizeof(bf16), stream);
  run_cell_seq(stream, Xg, whh_ef, EF.thr, SRC_, synT0, synT1, memT0, memT1,
               nullptr, memscr, spkseq);
  launch_gemm(stream, linout, linbf, spkseq, lin_ef, EF.lin_b,
              4096, H_, H_, H_, H_, H_, H_, 0, 0, nullptr, nullptr, nullptr, 0);
  launch_gemm(stream, mixbig, nullptr, linbf, mix_ef, nullptr,
              4096, H_, H_, H_, 1536, H_, H_, 0, 0, nullptr, nullptr, nullptr, 0);
  launch_gemm(stream, mixbig, fout, x_enc, mix_ef, EF.mix_b,
              4096, H_, E_, E_, 1536, H_, H_, H_, 3, EF.na, EF.nw, EF.nb, 0);

  // --- enc_b (reversed-time input) ---
  launch_gemm(stream, Xg, nullptr, x_rev, wih_eb, gb_eb,
              4096, G4_, E_, E_, E_, G4_, 0, 0, 0, nullptr, nullptr, nullptr, 0);
  hipMemsetAsync(synT0, 0, 16 * H_ * sizeof(float), stream);
  hipMemsetAsync(memT0, 0, 16 * H_ * sizeof(bf16), stream);
  run_cell_seq(stream, Xg, whh_eb, EB.thr, SRC_, synT0, synT1, memT0, memT1,
               nullptr, memscr, spkseq);
  launch_gemm(stream, linout, linbf, spkseq, lin_eb, EB.lin_b,
              4096, H_, H_, H_, H_, H_, H_, 0, 0, nullptr, nullptr, nullptr, 0);
  launch_gemm(stream, mixbig, nullptr, linbf, mix_eb, nullptr,
              4096, H_, H_, H_, 1536, H_, H_, 0, 0, nullptr, nullptr, nullptr, 0);
  launch_gemm(stream, mixbig, bout, x_rev, mix_eb, EB.mix_b,
              4096, H_, E_, E_, 1536, H_, H_, H_, 3, EB.na, EB.nw, EB.nb, 0);

  // --- concat [fwd, reversed-bwd] -> [S*16, 2H] ---
  k_concat_fb<<<dim3(SRC_, 16), 256, 0, stream>>>(cat2h, fout, bout, SRC_, H_);

  // --- enc layer 0 (in = 2H) ---
  launch_gemm(stream, Xg, nullptr, cat2h, wih_l0, gb_l0,
              4096, G4_, 2048, 2048, 2048, G4_, 0, 0, 0, nullptr, nullptr, nullptr, 0);
  hipMemsetAsync(synA0, 0, 16 * H_ * sizeof(float), stream);
  hipMemsetAsync(memA0, 0, 16 * H_ * sizeof(bf16), stream);
  int c0 = run_cell_seq(stream, Xg, whh_l0, L0.thr, SRC_, synA0, synA1,
                        memA0, memA1, mseq0, memscr, spkseq);
  launch_gemm(stream, linout, linbf, spkseq, lin_l0, L0.lin_b,
              4096, H_, H_, H_, H_, H_, H_, 0, 0, nullptr, nullptr, nullptr, 0);
  launch_gemm(stream, mixbig, nullptr, linbf, mix_l0, nullptr,
              4096, H_, H_, H_, 3072, H_, H_, 0, 0, nullptr, nullptr, nullptr, 0);
  launch_gemm(stream, mixbig, l0out, cat2h, mix_l0, L0.mix_b,
              4096, H_, 2048, 2048, 3072, H_, H_, H_, 3, L0.na, L0.nw, L0.nb, 0);

  // --- enc layer 1 (output only via states/mems; lin/mix are dead code) ---
  launch_gemm(stream, Xg, nullptr, l0out, wih_l1, gb_l1,
              4096, G4_, H_, H_, H_, G4_, 0, 0, 0, nullptr, nullptr, nullptr, 0);
  hipMemsetAsync(synB0, 0, 16 * H_ * sizeof(float), stream);
  hipMemsetAsync(memB0, 0, 16 * H_ * sizeof(bf16), stream);
  int c1 = run_cell_seq(stream, Xg, whh_l1, L1.thr, SRC_, synB0, synB1,
                        memB0, memB1, mseq1, memscr, spkseq);

  // ================================ decoder ================================
  k_embed_dyt<<<dim3(TGT_, 16), 256, 0, stream>>>(tx, dec_emb, tgt, dn_a, dn_w, dn_b, TGT_, E_);
  launch_gemm(stream, Xg, nullptr, tx, wih_d0, gb_d0,
              2048, G4_, E_, E_, E_, G4_, 0, 0, 0, nullptr, nullptr, nullptr, 0);

  float* sP0[2] = { synA0, synA1 }; bf16* mP0[2] = { memA0, memA1 };
  float* sP1[2] = { synB0, synB1 }; bf16* mP1[2] = { memB0, memB1 };

  for (int t = 0; t < TGT_; ++t) {
    // ---- layer 0 cell (precomputed Xg path) ----
    k_slstm_step<<<dim3(H_ / 32), 256, 0, stream>>>(
        Xg + (size_t)t * 16 * G4_, nullptr, nullptr, 0,
        whh_d0, nullptr, mP0[c0], sP0[c0], sP0[c0 ^ 1], memscr, mP0[c0 ^ 1],
        spk_st, D0.s.thr);
    c0 ^= 1;
    launch_gemm(stream, o_f, o_bf, spk_st, lin_d0, D0.s.lin_b,
                16, H_, H_, H_, H_, H_, H_, 0, 0, nullptr, nullptr, nullptr, 0);
    launch_gemm(stream, mixtmp, nullptr, o_bf, mix_d0, nullptr,
                16, H_, H_, H_, 1536, H_, H_, 0, 0, nullptr, nullptr, nullptr, 0);
    launch_gemm(stream, mixtmp, xcur0, tx + (size_t)t * 16 * E_, mix_d0, D0.s.mix_b,
                16, H_, E_, E_, 1536, H_, 2 * H_, H_, 3, D0.s.na, D0.s.nw, D0.s.nb, 0);
    launch_gemm(stream, q_f, nullptr, mP0[c0], attn_d0, nullptr,
                16, H_, H_, H_, H_, H_, H_, 0, 0, nullptr, nullptr, nullptr, 0);
    k_attention<<<dim3(16), 256, 0, stream>>>(xcur0 + H_, q_f, mseq0,
                                              D0.an_a, D0.an_w, D0.an_b);

    // ---- layer 1 cell (dynamic x = xcur0, in = 2H) ----
    k_slstm_step<<<dim3(H_ / 32), 256, 0, stream>>>(
        nullptr, xcur0, wih_d1, 2 * H_, whh_d1, gb_d1,
        mP1[c1], sP1[c1], sP1[c1 ^ 1], memscr, mP1[c1 ^ 1], spk_st, D1.s.thr);
    c1 ^= 1;
    bf16* xo = decout + (size_t)t * 16 * (2 * H_);
    launch_gemm(stream, o_f, o_bf, spk_st, lin_d1, D1.s.lin_b,
                16, H_, H_, H_, H_, H_, H_, 0, 0, nullptr, nullptr, nullptr, 0);
    launch_gemm(stream, mixtmp, nullptr, o_bf, mix_d1, nullptr,
                16, H_, H_, H_, 3072, H_, H_, 0, 0, nullptr, nullptr, nullptr, 0);
    launch_gemm(stream, mixtmp, xo, xcur0, mix_d1, D1.s.mix_b,
                16, H_, 2048, 2048, 3072, H_, 2 * H_, H_, 3, D1.s.na, D1.s.nw, D1.s.nb, 0);
    launch_gemm(stream, q_f, nullptr, mP1[c1], attn_d1, nullptr,
                16, H_, H_, H_, H_, H_, H_, 0, 0, nullptr, nullptr, nullptr, 0);
    k_attention<<<dim3(16), 256, 0, stream>>>(xo + H_, q_f, mseq1,
                                              D1.an_a, D1.an_w, D1.an_b);
  }

  // ---- logits: [2048, 2H] x [V, 2H]^T -> d_out[B,TGT,V] (row remap) ----
  launch_gemm(stream, out, nullptr, decout, fc_bf, fc_b,
              2048, V_, 2048, 2048, 2048, V_, 0, 0, 4,
              nullptr, nullptr, nullptr, TGT_);
}
